// Transformer_54760833024643
// MI455X (gfx1250) — matrix operand order repaired and checked by reading, not compiled, not run on hardware
//
#include <hip/hip_runtime.h>
#include <hip/hip_bf16.h>
#include <math.h>

// ---------------------------------------------------------------------------
// MI455X (gfx1250) implementation.
// Heavy math: N x N attention matmuls via v_wmma_f32_16x16x32_f16 (wave32).
// Softmax is over dim=1 (columns of the score matrix), so:
//   pass 1: colsum[m] = sum_n exp(s[n,m])
//   pass 2: out[n,c]  = t[n,c] + sum_m exp(s[n,m]) / colsum[m] * V[m,c]
// Scores are recomputed tile-wise (flash style) since the full 328MB score
// matrix exceeds L2 (192MB) and two HBM round-trips would cost more than
// recompute at f16 WMMA rates.  Softmax exp uses native v_exp_f32 (__expf):
// scores are O(+-5) so the native path is numerically safe, and it keeps the
// VALU side from throttling the WMMA pipe (libm expf is ~20 VALU ops/call).
// ---------------------------------------------------------------------------

typedef __attribute__((ext_vector_type(16))) _Float16 v16h;
typedef __attribute__((ext_vector_type(8)))  float    v8f;
typedef __attribute__((ext_vector_type(4)))  float    v4f;

#define NB       2
#define CH       128
#define HH       80
#define WW       80
#define NTOK     6400          // HH*WW
#define NTILES   400           // NTOK/16
#define TOKCH    819200        // NTOK*CH
#define SCALE_QK 0.08838834764831845f   // 1/sqrt(128)

__device__ __forceinline__ v8f wmma_f16(v16h a, v16h b, v8f c) {
  // v_wmma_f32_16x16x32_f16  (8-arg form: neg_a, A, neg_b, B, c_mod, C, reuse_a, reuse_b)
  return __builtin_amdgcn_wmma_f32_16x16x32_f16(false, a, false, b, (short)0, c, false, false);
}

// A-operand (16x32 f16, M x K), row-major source with given stride (halves).
// CDNA5 layout: lanes 0-15: row M=lane, halves0-7 = K0-7, halves8-15 = K16-23;
//               lanes 16-31: row M=lane-16, K8-15 / K24-31.
__device__ __forceinline__ v16h load_tileA(const _Float16* p0, int stride) {
  const int lane = threadIdx.x & 31;
  const int r    = lane & 15;
  const int kb   = (lane >> 4) << 3;     // 0 or 8
  const _Float16* p = p0 + r * stride + kb;
  v16h out;
  *((v4f*)&out)     = *(const v4f*)(p);        // halves 0..7  -> K kb..kb+7
  *((v4f*)&out + 1) = *(const v4f*)(p + 16);   // halves 8..15 -> K kb+16..kb+23
  return out;
}

// B-operand (32x16 f16, K x N). Source stored transposed: row index = column N,
// K contiguous within a row (stride in halves). Same K order as A operand:
// lanes 0-15 hold K0-7 / K16-23, lanes 16-31 hold K8-15 / K24-31.
__device__ __forceinline__ v16h load_tileB(const _Float16* p0, int stride) {
  const int lane = threadIdx.x & 31;
  const int r    = lane & 15;            // column N
  const int kb   = (lane >> 4) << 3;     // 0 or 8
  const _Float16* p = p0 + r * stride + kb;
  v16h out;
  *((v4f*)&out)     = *(const v4f*)(p);        // K kb..kb+7
  *((v4f*)&out + 1) = *(const v4f*)(p + 16);   // K kb+16..kb+23
  return out;
}

// ---------------------------------------------------------------------------
// Kernel 1: conv5x5(pad2) + BN(eval) + LeakyReLU(0.2) + 2D positional encoding
// Writes t[b][n][c] fp32.  One block (128 threads = 128 out-channels) per pixel.
// ---------------------------------------------------------------------------
__global__ void conv_pe_kernel(const float* __restrict__ x,
                               const float* __restrict__ cw,
                               const float* __restrict__ bng,
                               const float* __restrict__ bnb,
                               float* __restrict__ t) {
  const int pix = blockIdx.x;            // 0..12799
  const int b   = pix / NTOK;
  const int n   = pix - b * NTOK;
  const int h   = n / WW;
  const int w   = n - h * WW;
  const int tid = threadIdx.x;

  __shared__ float patch[232];           // 9 in-ch * 25 taps
  for (int idx = tid; idx < 225; idx += 128) {
    const int k   = idx / 25;            // contact channel = tt*3 + ci
    const int rem = idx - k * 25;
    const int kh  = rem / 5;
    const int kw  = rem - kh * 5;
    const int ci  = k % 3;
    const int tt  = k / 3;
    const int ih  = h + kh - 2;
    const int iw  = w + kw - 2;
    float v = 0.0f;
    if (ih >= 0 && ih < HH && iw >= 0 && iw < WW)
      v = x[((size_t)b * 57600) + (size_t)ci * 19200 + (size_t)tt * NTOK + ih * WW + iw];
    patch[idx] = v;
  }
  __syncthreads();

  const int oc = tid;
  const float* wp = cw + (size_t)oc * 225;
  float acc = 0.0f;
#pragma unroll 5
  for (int idx = 0; idx < 225; ++idx) acc = fmaf(patch[idx], wp[idx], acc);

  // BatchNorm eval (mean 0, var 1): y/sqrt(1+eps)*g + b, then LeakyReLU(0.2)
  float y = acc * 0.9999950000374997f * bng[oc] + bnb[oc];
  y = (y > 0.0f) ? y : 0.2f * y;

  // PositionalEncodingPermute2D, c=128, ch=64:
  //  c<32: sin(h*f[c]); 32..63: cos(h*f[c-32]); 64..95: sin(w*f[c-64]); 96..127: cos(w*f[c-96])
  const int  fi   = oc & 31;
  const float frq = expf(-(float)(2 * fi) * (9.210340371976184f / 64.0f)); // 10000^(-2i/64)
  const float pos = (oc < 64) ? (float)h : (float)w;
  const float ang = pos * frq;
  const float pe  = ((oc & 32) == 0) ? sinf(ang) : cosf(ang);

  t[(size_t)pix * CH + oc] = y + pe;
}

// ---------------------------------------------------------------------------
// Kernel 2: LayerNorm(ln1) + Q/K/V projections.
// Writes Qh[b][n][c] f16, Kh[b][n][c] f16, Vt[b][c][n] f16 (transposed!).
// One block (128 threads) per token.
// ---------------------------------------------------------------------------
__global__ void lnqkv_kernel(const float* __restrict__ t,
                             const float* __restrict__ g,
                             const float* __restrict__ bt,
                             const float* __restrict__ wq,
                             const float* __restrict__ wk,
                             const float* __restrict__ wv,
                             _Float16* __restrict__ Qh,
                             _Float16* __restrict__ Kh,
                             _Float16* __restrict__ Vt) {
  const int pix = blockIdx.x;
  const int b   = pix / NTOK;
  const int n   = pix - b * NTOK;
  const int tid = threadIdx.x;

  __shared__ float red[128];
  __shared__ float ln[128];

  const float xv = t[(size_t)pix * CH + tid];
  red[tid] = xv;
  __syncthreads();
  for (int s = 64; s > 0; s >>= 1) { if (tid < s) red[tid] += red[tid + s]; __syncthreads(); }
  const float mean = red[0] * (1.0f / 128.0f);
  __syncthreads();
  const float dv = xv - mean;
  red[tid] = dv * dv;
  __syncthreads();
  for (int s = 64; s > 0; s >>= 1) { if (tid < s) red[tid] += red[tid + s]; __syncthreads(); }
  const float var = red[0] * (1.0f / 128.0f);
  const float l   = dv * rsqrtf(var + 1e-5f) * g[tid] + bt[tid];
  ln[tid] = l;
  __syncthreads();

  float q = 0.0f, k = 0.0f, v = 0.0f;
#pragma unroll 4
  for (int i = 0; i < 128; ++i) {
    const float li = ln[i];
    q = fmaf(li, wq[i * 128 + tid], q);
    k = fmaf(li, wk[i * 128 + tid], k);
    v = fmaf(li, wv[i * 128 + tid], v);
  }
  Qh[(size_t)pix * CH + tid] = (_Float16)q;
  Kh[(size_t)pix * CH + tid] = (_Float16)k;
  Vt[(size_t)b * TOKCH + (size_t)tid * NTOK + n] = (_Float16)v;
}

// ---------------------------------------------------------------------------
// Kernel 3 (pass 1): colsum[m] = sum_n exp(s[n,m]/sqrt(c)).
// Block = 4 waves, one 16-column m-tile per block; waves split the n-tiles.
// K-tile B operands are loop-invariant and held in VGPRs.
// ---------------------------------------------------------------------------
__global__ void colsum_kernel(const _Float16* __restrict__ Qh,
                              const _Float16* __restrict__ Kh,
                              float* __restrict__ colsum) {
  const int b    = blockIdx.y;
  const int m0   = blockIdx.x * 16;
  const int wv   = threadIdx.x >> 5;
  const int lane = threadIdx.x & 31;
  const _Float16* Qb = Qh + (size_t)b * TOKCH;
  const _Float16* Kb = Kh + (size_t)b * TOKCH;

  v16h Kop[4];
#pragma unroll
  for (int kc = 0; kc < 4; ++kc)
    Kop[kc] = load_tileB(Kb + (size_t)m0 * CH + kc * 32, CH);

  float acc = 0.0f;
  for (int i = wv; i < NTILES; i += 4) {
    v8f d = {0.f, 0.f, 0.f, 0.f, 0.f, 0.f, 0.f, 0.f};
#pragma unroll
    for (int kc = 0; kc < 4; ++kc) {
      v16h a = load_tileA(Qb + (size_t)i * 16 * CH + kc * 32, CH);
      d = wmma_f16(a, Kop[kc], d);
    }
#pragma unroll
    for (int r = 0; r < 8; ++r) acc += __expf(d[r] * SCALE_QK);  // native v_exp_f32
  }
  // lanes L and L+16 cover disjoint n-rows of the same column (lane&15)
  acc += __shfl_xor(acc, 16);

  __shared__ float sred[64];
  if (lane < 16) sred[wv * 16 + lane] = acc;
  __syncthreads();
  if (threadIdx.x < 16) {
    const float s = sred[threadIdx.x] + sred[16 + threadIdx.x] +
                    sred[32 + threadIdx.x] + sred[48 + threadIdx.x];
    colsum[b * NTOK + m0 + threadIdx.x] = s;
  }
}

// ---------------------------------------------------------------------------
// Kernel 4 (pass 2): attn[n,c] = t[n,c] + sum_m exp(s[n,m])/colsum[m] * V[m,c]
// Block = 4 waves, one 16-row n-tile. Each iteration covers 128 m:
//  - wave w recomputes its own 16x32 P chunk (8 wmma), writes it to LDS (f16)
//  - after barrier, every wave applies all 4 P chunks to its 32-wide c-slice
//    of V (8 wmma).  Q A-operands are loop-invariant and held in VGPRs.
// ---------------------------------------------------------------------------
__global__ void attn_kernel(const _Float16* __restrict__ Qh,
                            const _Float16* __restrict__ Kh,
                            const _Float16* __restrict__ Vt,
                            const float* __restrict__ colsum,
                            const float* __restrict__ t,
                            float* __restrict__ attn) {
  const int b    = blockIdx.y;
  const int n0   = blockIdx.x * 16;
  const int wv   = threadIdx.x >> 5;
  const int lane = threadIdx.x & 31;
  const _Float16* Qb = Qh + (size_t)b * TOKCH;
  const _Float16* Kb = Kh + (size_t)b * TOKCH;
  const _Float16* Vb = Vt + (size_t)b * TOKCH;
  const float*    cs = colsum + b * NTOK;

  __shared__ alignas(16) _Float16 P[4][16 * 32];

  v16h Qa[4];
#pragma unroll
  for (int kc = 0; kc < 4; ++kc)
    Qa[kc] = load_tileA(Qb + (size_t)n0 * CH + kc * 32, CH);

  v8f acc0 = {0.f, 0.f, 0.f, 0.f, 0.f, 0.f, 0.f, 0.f};
  v8f acc1 = {0.f, 0.f, 0.f, 0.f, 0.f, 0.f, 0.f, 0.f};
  const int cb = wv * 32;

  for (int it = 0; it < 50; ++it) {
    const int mbase = it * 128;
    const int mch   = mbase + wv * 32;

    // --- recompute this wave's 16x32 probability chunk into LDS ---
#pragma unroll
    for (int half = 0; half < 2; ++half) {
      const int mrow = mch + half * 16;
      v8f d = {0.f, 0.f, 0.f, 0.f, 0.f, 0.f, 0.f, 0.f};
#pragma unroll
      for (int kc = 0; kc < 4; ++kc) {
        v16h bo = load_tileB(Kb + (size_t)mrow * CH + kc * 32, CH);
        d = wmma_f16(Qa[kc], bo, d);
      }
      const float rc = 1.0f / cs[mrow + (lane & 15)];
#pragma unroll
      for (int r = 0; r < 8; ++r) {
        const float p  = __expf(d[r] * SCALE_QK) * rc;     // native v_exp_f32
        const int   nl = r + ((lane >> 4) << 3);           // D layout: M = r + 8*(lane/16)
        P[wv][nl * 32 + half * 16 + (lane & 15)] = (_Float16)p;
      }
    }
    __syncthreads();

    // --- apply all 4 P chunks to this wave's 32-wide c-slice of V ---
#pragma unroll
    for (int j = 0; j < 4; ++j) {
      v16h a  = load_tileA(&P[j][0], 32);
      v16h b0 = load_tileB(Vb + (size_t)cb * NTOK + mbase + j * 32, NTOK);
      acc0 = wmma_f16(a, b0, acc0);
      v16h b1 = load_tileB(Vb + (size_t)(cb + 16) * NTOK + mbase + j * 32, NTOK);
      acc1 = wmma_f16(a, b1, acc1);
    }
    __syncthreads();
  }

  // epilogue: residual add, write attn[b][n][c] fp32
  const size_t base = ((size_t)b * NTOK + n0) * CH;
#pragma unroll
  for (int r = 0; r < 8; ++r) {
    const int nl = r + ((lane >> 4) << 3);
    const int c0 = cb + (lane & 15);
    const size_t i0 = base + (size_t)nl * CH + c0;
    attn[i0] = t[i0] + acc0[r];
    const size_t i1 = i0 + 16;
    attn[i1] = t[i1] + acc1[r];
  }
}

// ---------------------------------------------------------------------------
// Kernel 5: LayerNorm(ln2) + MLP (gelu exact) + residual; transpose to NCHW out.
// One block (128 threads) per token.
// ---------------------------------------------------------------------------
__global__ void mlp_kernel(const float* __restrict__ attn,
                           const float* __restrict__ g,
                           const float* __restrict__ bt,
                           const float* __restrict__ fc1w,
                           const float* __restrict__ fc1b,
                           const float* __restrict__ fc2w,
                           const float* __restrict__ fc2b,
                           float* __restrict__ out) {
  const int pix = blockIdx.x;
  const int b   = pix / NTOK;
  const int n   = pix - b * NTOK;
  const int tid = threadIdx.x;

  __shared__ float red[128];
  __shared__ float hid[128];
  __shared__ float m1[256];

  const float xv = attn[(size_t)pix * CH + tid];
  red[tid] = xv;
  __syncthreads();
  for (int s = 64; s > 0; s >>= 1) { if (tid < s) red[tid] += red[tid + s]; __syncthreads(); }
  const float mean = red[0] * (1.0f / 128.0f);
  __syncthreads();
  const float dv = xv - mean;
  red[tid] = dv * dv;
  __syncthreads();
  for (int s = 64; s > 0; s >>= 1) { if (tid < s) red[tid] += red[tid + s]; __syncthreads(); }
  const float var = red[0] * (1.0f / 128.0f);
  hid[tid] = dv * rsqrtf(var + 1e-5f) * g[tid] + bt[tid];
  __syncthreads();

#pragma unroll
  for (int half = 0; half < 2; ++half) {
    const int o = tid + half * 128;
    float s = fc1b[o];
#pragma unroll 4
    for (int i = 0; i < 128; ++i) s = fmaf(hid[i], fc1w[i * 256 + o], s);
    // exact GELU: 0.5*x*(1+erf(x/sqrt(2)))
    m1[o] = 0.5f * s * (1.0f + erff(s * 0.70710678118654752f));
  }
  __syncthreads();

  float s2 = fc2b[tid];
#pragma unroll 4
  for (int i = 0; i < 256; ++i) s2 = fmaf(m1[i], fc2w[i * 128 + tid], s2);

  out[(size_t)b * TOKCH + (size_t)tid * NTOK + n] = xv + s2;
}

// ---------------------------------------------------------------------------
extern "C" void kernel_launch(void* const* d_in, const int* in_sizes, int n_in,
                              void* d_out, int out_size, void* d_ws, size_t ws_size,
                              hipStream_t stream) {
  (void)in_sizes; (void)n_in; (void)out_size; (void)ws_size;

  const float* x      = (const float*)d_in[0];
  const float* conv0w = (const float*)d_in[1];
  const float* bn_g   = (const float*)d_in[2];
  const float* bn_b   = (const float*)d_in[3];
  const float* ln1_g  = (const float*)d_in[4];
  const float* ln1_b  = (const float*)d_in[5];
  const float* wq     = (const float*)d_in[6];
  const float* wk     = (const float*)d_in[7];
  const float* wv     = (const float*)d_in[8];
  const float* ln2_g  = (const float*)d_in[9];
  const float* ln2_b  = (const float*)d_in[10];
  const float* fc1_w  = (const float*)d_in[11];
  const float* fc1_b  = (const float*)d_in[12];
  const float* fc2_w  = (const float*)d_in[13];
  const float* fc2_b  = (const float*)d_in[14];
  float* out = (float*)d_out;

  // workspace layout (fp32 / f16 regions, all 256B aligned)
  float*    t_ws    = (float*)d_ws;                 // 2*6400*128 fp32
  float*    attn_ws = t_ws + 2 * TOKCH;             // 2*6400*128 fp32
  float*    cs_ws   = attn_ws + 2 * TOKCH;          // 2*6400    fp32
  _Float16* Qh      = (_Float16*)(cs_ws + 2 * NTOK);
  _Float16* Kh      = Qh + 2 * TOKCH;
  _Float16* Vt      = Kh + 2 * TOKCH;

  const int npix = NB * NTOK;                        // 12800

  conv_pe_kernel<<<npix, 128, 0, stream>>>(x, conv0w, bn_g, bn_b, t_ws);
  lnqkv_kernel<<<npix, 128, 0, stream>>>(t_ws, ln1_g, ln1_b, wq, wk, wv, Qh, Kh, Vt);
  colsum_kernel<<<dim3(NTILES, NB), 128, 0, stream>>>(Qh, Kh, cs_ws);
  attn_kernel<<<dim3(NTILES, NB), 128, 0, stream>>>(Qh, Kh, Vt, cs_ws, t_ws, attn_ws);
  mlp_kernel<<<npix, 128, 0, stream>>>(attn_ws, ln2_g, ln2_b, fc1_w, fc1_b, fc2_w, fc2_b, out);
}
